// QLSTMClassifier_47390669144591
// MI455X (gfx1250) — matrix operand order repaired and checked by reading, not compiled, not run on hardware
//
#include <hip/hip_runtime.h>
#include <hip/hip_bf16.h>

#define BB     512
#define SS     128
#define EMBD   256
#define HIDD   256
#define NWIRE  8
#define NSTATE 256
#define NCLS   4
#define RSTRIDE 264   // 256 + 8 halves padding (LDS bank-conflict mitigation)
#define FSTRIDE 260   // float-row stride for state arrays
#define WSTRIDE 264   // float-row stride for W_in LDS copy

typedef _Float16 v16h __attribute__((ext_vector_type(16)));
typedef _Float16 v8h  __attribute__((ext_vector_type(8)));
typedef float    v8f  __attribute__((ext_vector_type(8)));

__device__ __forceinline__ float sigm(float x) { return 1.0f / (1.0f + __expf(-x)); }
__device__ __forceinline__ float tanh_fast(float x) {
    x = fminf(fmaxf(x, -15.0f), 15.0f);
    float e = __expf(2.0f * x);
    return (e - 1.0f) / (e + 1.0f);
}
__device__ __forceinline__ float wave_sum(float s) {   // butterfly: full sum in every lane
#pragma unroll
    for (int o = 16; o > 0; o >>= 1) s += __shfl_xor(s, o, 32);
    return s;
}

// A-fragment (16x32 f16): lanes 0-15 hold K {0..7, 16..23}; lanes 16-31 hold K {8..15, 24..31}
__device__ __forceinline__ v16h load_a_frag(const _Float16* rowp, int k0, int lane) {
    int kh = (lane >> 4) * 8;
    v8h lo = *(const v8h*)(rowp + k0 + kh);
    v8h hi = *(const v8h*)(rowp + k0 + kh + 16);
    return __builtin_shufflevector(lo, hi, 0,1,2,3,4,5,6,7,8,9,10,11,12,13,14,15);
}
// B-fragment (32x16 f16): lanes 0-15 hold K {0..7, 16..23}; lanes 16-31 hold K {8..15, 24..31}
__device__ __forceinline__ v16h load_b_frag(const _Float16* colp, int k0, int lane) {
    int kb = (lane >> 4) * 8;
    v8h lo = *(const v8h*)(colp + k0 + kb);
    v8h hi = *(const v8h*)(colp + k0 + kb + 16);
    return __builtin_shufflevector(lo, hi, 0,1,2,3,4,5,6,7,8,9,10,11,12,13,14,15);
}

// async global->LDS copy of 16 bytes per active lane (CDNA5 ASYNCcnt path)
__device__ __forceinline__ void async_copy_b128(void* lds_dst, const void* gsrc) {
    unsigned ldsoff = (unsigned)(uintptr_t)lds_dst;   // low 32 bits of generic LDS ptr = LDS offset
    asm volatile("global_load_async_to_lds_b128 %0, %1, off"
                 :: "v"(ldsoff), "v"(gsrc) : "memory");
}
__device__ __forceinline__ void async_wait0() {
    asm volatile("s_wait_asynccnt 0x0" ::: "memory");
}

// -------- xW[t,b,w] = x[b,t,:] . W_in[w, HID:HID+EMB]  (one pass over 64MB, bandwidth bound)
__global__ void k_xw(const float* __restrict__ x, const float* __restrict__ W_in,
                     float* __restrict__ xW) {
    int t = blockIdx.x, b = blockIdx.y, tid = threadIdx.x;
    __shared__ float xl[EMBD];
    xl[tid] = x[((size_t)b * SS + t) * EMBD + tid];
    __syncthreads();
    int w = tid >> 5, lane = tid & 31;
    const float* wr = W_in + w * (EMBD + HIDD) + HIDD;
    float s = 0.0f;
#pragma unroll
    for (int i = 0; i < EMBD / 32; i++) s += xl[lane + 32 * i] * wr[lane + 32 * i];
    s = wave_sum(s);
    if (lane == 0) xW[((size_t)t * BB + b) * NWIRE + w] = s;
}

// -------- build Utilde_g = U_fixed(wq_g) * diag((-i)^popcnt(k)), re/im rows interleaved, f16
__global__ void k_build_u(const float* wq0, const float* wq1, const float* wq2, const float* wq3,
                          _Float16* __restrict__ U) {
    int k = blockIdx.x, g = blockIdx.y, tid = threadIdx.x;
    const float* wq = (g == 0) ? wq0 : (g == 1) ? wq1 : (g == 2) ? wq2 : wq3;
    __shared__ float sre[NSTATE], smi[NSTATE];
    sre[tid] = (tid == k) ? 1.0f : 0.0f;
    smi[tid] = 0.0f;
    __syncthreads();
    for (int l = 0; l < 2; l++) {
        for (int i = 0; i < NWIRE; i++) {  // RX(wq[l][i]) on wire i
            float th = wq[l * NWIRE + i];
            float c = cosf(0.5f * th), s = sinf(0.5f * th);
            int bp = NWIRE - 1 - i;
            if (tid < NSTATE / 2) {
                int lm = (1 << bp) - 1;
                int i0 = ((tid & ~lm) << 1) | (tid & lm);
                int i1 = i0 | (1 << bp);
                float x0 = sre[i0], y0 = smi[i0], x1 = sre[i1], y1 = smi[i1];
                sre[i0] = c * x0 + s * y1;  smi[i0] = c * y0 - s * x1;
                sre[i1] = c * x1 + s * y0;  smi[i1] = c * y1 - s * x0;
            }
            __syncthreads();
        }
        for (int i = 0; i < NWIRE; i++) {  // CNOT(i -> i+1 mod 8)
            int bc = NWIRE - 1 - i;
            int bt2 = NWIRE - 1 - ((i + 1) & (NWIRE - 1));
            if ((((tid >> bc) & 1) == 1) && (((tid >> bt2) & 1) == 0)) {
                int p = tid | (1 << bt2);
                float tr = sre[tid], ti = smi[tid];
                sre[tid] = sre[p]; smi[tid] = smi[p];
                sre[p] = tr;       smi[p] = ti;
            }
            __syncthreads();
        }
    }
    int pop = __popc(k) & 3;
    float pr = (pop == 0) ? 1.0f : ((pop == 2) ? -1.0f : 0.0f);
    float pi = (pop == 1) ? -1.0f : ((pop == 3) ? 1.0f : 0.0f);
    float re = sre[tid] * pr - smi[tid] * pi;
    float im = sre[tid] * pi + smi[tid] * pr;
    _Float16* Ug = U + (size_t)g * 2 * NSTATE * NSTATE;
    Ug[(size_t)(2 * tid) * NSTATE + k]     = (_Float16)re;
    Ug[(size_t)(2 * tid + 1) * NSTATE + k] = (_Float16)im;
}

// -------- M_comb[h,k] = sum_w W_out[h,w] * sign_w(k)
__global__ void k_build_mc(const float* __restrict__ W_out, _Float16* __restrict__ Mc) {
    int h = blockIdx.x, k = threadIdx.x;
    float s = 0.0f;
#pragma unroll
    for (int w = 0; w < NWIRE; w++) {
        float sg = ((k >> (NWIRE - 1 - w)) & 1) ? -1.0f : 1.0f;
        s += W_out[h * NWIRE + w] * sg;
    }
    Mc[(size_t)h * NSTATE + k] = (_Float16)s;
}

// ======== persistent fused scan: 32 blocks x 512 threads, 16 batch elements/block ========
__global__ void __launch_bounds__(512) k_qlstm(
        const float* __restrict__ xW, const _Float16* __restrict__ U,
        const _Float16* __restrict__ Mc, const float* __restrict__ W_in,
        const float* __restrict__ b_in, const float* __restrict__ b_out,
        const int* __restrict__ seqlen, const float* __restrict__ W_cls,
        const float* __restrict__ b_cls, float* __restrict__ out) {
    __shared__ __align__(16) _Float16 Rl[16 * RSTRIDE];        // embedded product states
    __shared__ __align__(16) _Float16 Pl[4][16 * RSTRIDE];     // probabilities per gate
    __shared__ float hs[16 * FSTRIDE];                         // h state
    __shared__ float csb[16 * FSTRIDE];                        // c state
    __shared__ float lsb[16 * FSTRIDE];                        // gathered last-h
    __shared__ float whl[8 * WSTRIDE];                         // W_in h-part
    __shared__ float bol[256];
    __shared__ float bil[8];
    __shared__ __align__(16) float xwl[2][128];                // double-buffered xW[t] tile

    int bt = blockIdx.x, tid = threadIdx.x;
    int wv = tid >> 5, lane = tid & 31;
    int bl = lane & 15, lh = lane >> 4;

    // ---- per-block init (state lives in LDS for the whole scan) ----
    for (int i = tid; i < 16 * FSTRIDE; i += 512) { hs[i] = 0.f; csb[i] = 0.f; lsb[i] = 0.f; }
    for (int i = tid; i < 8 * 256; i += 512) {
        int w8 = i >> 8, e = i & 255;
        whl[w8 * WSTRIDE + e] = W_in[w8 * (EMBD + HIDD) + e];
    }
    if (tid < 256) bol[tid] = b_out[tid];
    if (tid < 8)   bil[tid] = b_in[tid];
    int slv = seqlen[bt * 16 + bl]; if (slv < 1) slv = 1;
    int tsel = slv - 1;

    if (tid < 32)  // async prefetch of xW tile for t=0
        async_copy_b128(&xwl[0][lane * 4], xW + ((size_t)bt * 16) * NWIRE + lane * 4);
    __syncthreads();

#pragma unroll 1
    for (int t = 0; t < SS; t++) {
        if (tid < 32) async_wait0();
        __syncthreads();   // B1: xwl[t&1] arrived; hs/cs from previous step visible

        // ---- phase 1: angles + embedded state R, wave wv handles batch-local b = wv ----
        {
            int b = wv;
            const float* hrow = hs + b * FSTRIDE;
            float cv[8], sv[8];
#pragma unroll
            for (int w8 = 0; w8 < 8; w8++) {
                const float* wrow = whl + w8 * WSTRIDE;
                float s = 0.0f;
#pragma unroll
                for (int i = 0; i < 8; i++) s += hrow[lane + 32 * i] * wrow[lane + 32 * i];
                s = wave_sum(s);
                float y = s + bil[w8] + xwl[t & 1][b * 8 + w8];
                float hy = 0.5f * y;
                cv[w8] = __cosf(hy);
                sv[w8] = __sinf(hy);
            }
#pragma unroll
            for (int i = 0; i < 8; i++) {
                int k = lane + 32 * i;
                float r = 1.0f;
#pragma unroll
                for (int j = 0; j < 8; j++) r *= ((k >> (7 - j)) & 1) ? sv[j] : cv[j];
                Rl[b * RSTRIDE + k] = (_Float16)r;
            }
        }
        __syncthreads();   // B2: Rl ready

        // ---- phase 2: C = Utilde_g @ R, P = |C|^2 -> LDS; 128 tiles over 16 waves ----
        {
            const _Float16* bcol = Rl + bl * RSTRIDE;   // loop-invariant B fragments
#pragma unroll 1
            for (int ti = wv * 8; ti < wv * 8 + 8; ti++) {
                int g = ti >> 5, mt = ti & 31;
                const _Float16* arow = U + (size_t)g * 2 * NSTATE * NSTATE
                                         + (size_t)(mt * 16 + bl) * NSTATE;
                __builtin_prefetch(arow + 16 * NSTATE, 0, 1);
                v8f acc = {0.f, 0.f, 0.f, 0.f, 0.f, 0.f, 0.f, 0.f};
#pragma unroll
                for (int kk = 0; kk < 8; kk++) {
                    v16h a  = load_a_frag(arow, kk * 32, lane);
                    v16h bf = load_b_frag(bcol, kk * 32, lane);
                    acc = __builtin_amdgcn_wmma_f32_16x16x32_f16(false, a, false, bf,
                                                                 (short)0, acc, false, false);
                }
                int kbase = mt * 8 + lh * 4;   // rows re/im interleaved -> 4 probs per lane
#pragma unroll
                for (int j = 0; j < 4; j++) {
                    float p = acc[2 * j] * acc[2 * j] + acc[2 * j + 1] * acc[2 * j + 1];
                    Pl[g][bl * RSTRIDE + kbase + j] = (_Float16)p;
                }
            }
        }
        __syncthreads();   // B3: Pl ready

        // ---- phase 3: gate GEMM (all 4 gates in one wave) + in-register LSTM ----
        {
            int ht = wv;
            const _Float16* arow = Mc + (size_t)(ht * 16 + bl) * NSTATE;
            const _Float16* bc0 = Pl[0] + bl * RSTRIDE;
            const _Float16* bc1 = Pl[1] + bl * RSTRIDE;
            const _Float16* bc2 = Pl[2] + bl * RSTRIDE;
            const _Float16* bc3 = Pl[3] + bl * RSTRIDE;
            v8f ac0 = {0.f,0.f,0.f,0.f,0.f,0.f,0.f,0.f};
            v8f ac1 = ac0, ac2 = ac0, ac3 = ac0;
#pragma unroll
            for (int kk = 0; kk < 8; kk++) {
                v16h a = load_a_frag(arow, kk * 32, lane);   // shared A across the 4 gates
                v16h b0 = load_b_frag(bc0, kk * 32, lane);
                ac0 = __builtin_amdgcn_wmma_f32_16x16x32_f16(false, a, false, b0, (short)0, ac0, false, false);
                v16h b1 = load_b_frag(bc1, kk * 32, lane);
                ac1 = __builtin_amdgcn_wmma_f32_16x16x32_f16(false, a, false, b1, (short)0, ac1, false, false);
                v16h b2 = load_b_frag(bc2, kk * 32, lane);
                ac2 = __builtin_amdgcn_wmma_f32_16x16x32_f16(false, a, false, b2, (short)0, ac2, false, false);
                v16h b3 = load_b_frag(bc3, kk * 32, lane);
                ac3 = __builtin_amdgcn_wmma_f32_16x16x32_f16(false, a, false, b3, (short)0, ac3, false, false);
            }
            // overlap: kick off async xW prefetch for t+1 while doing the LSTM math
            if (t + 1 < SS && tid < 32)
                async_copy_b128(&xwl[(t + 1) & 1][lane * 4],
                                xW + ((size_t)(t + 1) * BB + bt * 16) * NWIRE + lane * 4);
#pragma unroll
            for (int j = 0; j < 8; j++) {
                int hh = ht * 16 + j + lh * 8;
                float bo = bol[hh];
                float f  = sigm(ac0[j] + bo);
                float ii = sigm(ac1[j] + bo);
                float gg = tanh_fast(ac2[j] + bo);
                float oo = sigm(ac3[j] + bo);
                int ix = bl * FSTRIDE + hh;
                float cn = f * csb[ix] + ii * gg;
                float hn = oo * tanh_fast(cn);
                csb[ix] = cn;
                hs[ix]  = hn;
                if (t == tsel) lsb[ix] = hn;
            }
        }
        // next iteration's B1 separates phase-3 writes from phase-1 reads
    }
    __syncthreads();

    // ---- tail: classifier softmax (wave per batch element) + h/c copy-out ----
    {
        int b = wv, b2 = bt * 16 + b;
        float lg[NCLS];
#pragma unroll
        for (int j = 0; j < NCLS; j++) {
            const float* wc = W_cls + j * HIDD;
            float s = 0.0f;
#pragma unroll
            for (int i = 0; i < 8; i++) s += lsb[b * FSTRIDE + lane + 32 * i] * wc[lane + 32 * i];
            s = wave_sum(s);
            lg[j] = s + b_cls[j];
        }
        if (lane == 0) {
            float m = fmaxf(fmaxf(lg[0], lg[1]), fmaxf(lg[2], lg[3]));
            float e0 = __expf(lg[0] - m), e1 = __expf(lg[1] - m);
            float e2 = __expf(lg[2] - m), e3 = __expf(lg[3] - m);
            float inv = 1.0f / (e0 + e1 + e2 + e3);
            out[(size_t)b2 * NCLS + 0] = e0 * inv;
            out[(size_t)b2 * NCLS + 1] = e1 * inv;
            out[(size_t)b2 * NCLS + 2] = e2 * inv;
            out[(size_t)b2 * NCLS + 3] = e3 * inv;
        }
        float* oh = out + (size_t)BB * NCLS;
        float* oc = oh + (size_t)BB * HIDD;
#pragma unroll
        for (int i = 0; i < 8; i++) {
            int e = lane + 32 * i;
            oh[(size_t)b2 * HIDD + e] = hs[b * FSTRIDE + e];
            oc[(size_t)b2 * HIDD + e] = csb[b * FSTRIDE + e];
        }
    }
}

extern "C" void kernel_launch(void* const* d_in, const int* in_sizes, int n_in,
                              void* d_out, int out_size, void* d_ws, size_t ws_size,
                              hipStream_t stream) {
    const float* x     = (const float*)d_in[0];
    const int*   seqln = (const int*)  d_in[1];
    const float* W_in  = (const float*)d_in[2];
    const float* b_in  = (const float*)d_in[3];
    const float* wqf   = (const float*)d_in[4];
    const float* wqi   = (const float*)d_in[5];
    const float* wqu   = (const float*)d_in[6];
    const float* wqo   = (const float*)d_in[7];
    const float* W_out = (const float*)d_in[8];
    const float* b_out = (const float*)d_in[9];
    const float* W_cls = (const float*)d_in[10];
    const float* b_cls = (const float*)d_in[11];
    float* out = (float*)d_out;

    char* ws = (char*)d_ws;
    size_t off = 0;
    float*    xW = (float*)(ws + off);    off += (size_t)SS * BB * NWIRE * 4;         // 2 MB
    _Float16* U  = (_Float16*)(ws + off); off += (size_t)4 * 2 * NSTATE * NSTATE * 2; // 1 MB
    _Float16* Mc = (_Float16*)(ws + off); off += (size_t)NSTATE * NSTATE * 2;         // 128 KB

    k_xw<<<dim3(SS, BB), 256, 0, stream>>>(x, W_in, xW);
    k_build_u<<<dim3(NSTATE, 4), NSTATE, 0, stream>>>(wqf, wqi, wqu, wqo, U);
    k_build_mc<<<NSTATE, NSTATE, 0, stream>>>(W_out, Mc);
    k_qlstm<<<BB / 16, 512, 0, stream>>>(xW, U, Mc, W_in, b_in, b_out, seqln,
                                         W_cls, b_cls, out);
}
